// CompressAttention_54039278518856
// MI455X (gfx1250) — compile-verified, matrix operand order checked by reading, not run on hardware
//
#include <hip/hip_runtime.h>

typedef __attribute__((ext_vector_type(16))) __bf16 v16bf;
typedef __attribute__((ext_vector_type(8)))  __bf16 v8bf;
typedef __attribute__((ext_vector_type(8)))  float  v8f;

#define NUM_AREAS 500
#define NUM_HEADS 4
#define LL    200      // area length
#define LP    208      // padded rows (13*16)
#define DD    32       // head depth
#define DM    128      // d_model
#define RELV  399      // 2L-1 valid relative positions
#define RELP  400      // padded (25*16)
#define MPAD  224      // padded key cols for P@v (7*32)
#define NWAVE 4
#define SCALE    0.08838834764831843f   /* 1/sqrt(128) */
#define MASKNEG  (-6550.4f)             /* fp16 min / 10 */

// --- LDS layout (byte offsets), phase-1 region overlaid by phase-2 bands ---
constexpr int SM_QS  = 0;                       // LP*DD bf16     = 13312
constexpr int SM_KS  = SM_QS + LP * DD * 2;     // 13312
constexpr int SM_VT  = SM_KS + LP * DD * 2;     // 26624  (DD*MPAD bf16)
constexpr int SM_EU  = SM_VT + DD * MPAD * 2;   // 40960  (RELP*DD bf16)
constexpr int SM_OVL = SM_EU + RELP * DD * 2;   // 66560
// phase 1 overlay: x tile + per-head weight slices + bias
constexpr int SM_XS  = SM_OVL;                  // LP*DM bf16     = 53248
constexpr int SM_WT  = SM_XS + LP * DM * 2;     // 3*DD*DM bf16   = 24576
constexpr int SM_BI  = SM_WT + 3 * DD * DM * 2; // 96*4
constexpr int SM_END1 = SM_BI + 96 * 4;         // 144768
// phase 2 overlay: T band (16 x RELP f32), S band (16 x MPAD f32), reductions
constexpr int SM_TB  = SM_OVL;                  // 25600
constexpr int SM_SB  = SM_TB + 16 * RELP * 4;   // 14336
constexpr int SM_RED = SM_SB + 16 * MPAD * 4;   // 32*4
constexpr int SM_END2 = SM_RED + 32 * 4;
constexpr int SM_BYTES = (SM_END1 > SM_END2) ? SM_END1 : SM_END2;

// Load a 16x32 bf16 WMMA operand fragment from row-major storage.
// A-matrix: row = lane&15, K runs {ko..ko+7, ko+16..ko+23}, ko = (lane&16)?8:0.
// B-matrix uses the same addressing on N-major (i.e. transposed) storage.
__device__ __forceinline__ v16bf frag_bf(const __bf16* base, int stride, int lane) {
  const __bf16* p = base + (lane & 15) * stride + ((lane & 16) ? 8 : 0);
  v8bf lo = *(const v8bf*)(p);
  v8bf hi = *(const v8bf*)(p + 16);
  v16bf r;
#pragma unroll
  for (int i = 0; i < 8; ++i) { r[i] = lo[i]; r[8 + i] = hi[i]; }
  return r;
}

__device__ __forceinline__ v16bf frag_f32(const float* base, int stride, int lane) {
  const float* p = base + (lane & 15) * stride + ((lane & 16) ? 8 : 0);
  v16bf r;
#pragma unroll
  for (int i = 0; i < 8; ++i) { r[i] = (__bf16)p[i]; r[8 + i] = (__bf16)p[16 + i]; }
  return r;
}

__device__ __forceinline__ v8f wmma_bf16(v16bf a, v16bf b, v8f c) {
  return __builtin_amdgcn_wmma_f32_16x16x32_bf16(false, a, false, b, (short)0, c,
                                                 false, false);
}

__global__ __launch_bounds__(128) void compress_attn_kernel(
    const float* __restrict__ inp,   // (100000,128)
    const float* __restrict__ mask,  // (200,200)
    const float* __restrict__ Wq, const float* __restrict__ bq,
    const float* __restrict__ Wk, const float* __restrict__ bk,
    const float* __restrict__ Wv, const float* __restrict__ bv,
    const float* __restrict__ E,     // (402,32)
    float* __restrict__ out,         // (500,200,128)
    float* __restrict__ attn)        // (500,4,200,200)
{
  __shared__ alignas(16) char smem[SM_BYTES];
  __bf16* qs   = (__bf16*)(smem + SM_QS);   // [LP][DD], pre-scaled by 1/sqrt(128)
  __bf16* ks   = (__bf16*)(smem + SM_KS);   // [LP][DD]
  __bf16* vT   = (__bf16*)(smem + SM_VT);   // [DD][MPAD] (transposed v)
  __bf16* Eus  = (__bf16*)(smem + SM_EU);   // [RELP][DD]
  __bf16* xs   = (__bf16*)(smem + SM_XS);   // [LP][DM]
  __bf16* wt   = (__bf16*)(smem + SM_WT);   // [3][DD][DM]  (W transposed: [e][d])
  float*  biasS= (float*)(smem + SM_BI);    // [3][DD]
  float*  Tband= (float*)(smem + SM_TB);    // [16][RELP]
  float*  Sband= (float*)(smem + SM_SB);    // [16][MPAD]
  float*  red  = (float*)(smem + SM_RED);   // [0..15]=rowmax, [16..31]=inv rowsum

  const int blk  = blockIdx.x;
  const int a    = blk >> 2;
  const int h    = blk & 3;
  const int tid  = threadIdx.x;
  const int wave = tid >> 5;
  const int lane = tid & 31;
  const float* W3[3] = { Wq, Wk, Wv };
  const float* B3[3] = { bq, bk, bv };

  // ---------------- Phase 0: stage inputs into LDS ----------------
  for (int idx = tid; idx < RELP * DD; idx += 128) {            // Eu = E[3:402], pad row 399
    int m = idx >> 5, d = idx & 31;
    Eus[idx] = (m < RELV) ? (__bf16)E[(3 + m) * DD + d] : (__bf16)0.f;
  }
  for (int idx = tid; idx < LP * DM; idx += 128) {              // x tile, zero-padded rows
    int l = idx >> 7, d = idx & 127;
    xs[idx] = (l < LL) ? (__bf16)inp[(a * LL + l) * DM + d] : (__bf16)0.f;
  }
  for (int idx = tid; idx < 3 * DD * DM; idx += 128) {          // W columns for this head, transposed
    int o = idx / (DD * DM), r = idx % (DD * DM);
    int e = r >> 7, d = r & 127;
    wt[idx] = (__bf16)W3[o][d * DM + h * DD + e];
  }
  if (tid < 96) { int o = tid >> 5, e = tid & 31; biasS[tid] = B3[o][h * DD + e]; }
  for (int idx = tid; idx < DD * 16; idx += 128) {              // zero v pad keys 208..223
    int d = idx >> 4, j = 208 + (idx & 15);
    vT[d * MPAD + j] = (__bf16)0.f;
  }
  __syncthreads();

  // ---------------- Phase 1: q/k/v projection via WMMA ----------------
  for (int mi = wave; mi < 13; mi += NWAVE) {
    v16bf af[4];
#pragma unroll
    for (int kk = 0; kk < 4; ++kk) af[kk] = frag_bf(xs + mi * 16 * DM + kk * 32, DM, lane);
#pragma unroll
    for (int o = 0; o < 3; ++o) {
#pragma unroll
      for (int nt = 0; nt < 2; ++nt) {
        v8f acc = {};
#pragma unroll
        for (int kk = 0; kk < 4; ++kk) {
          v16bf bf = frag_bf(wt + o * DD * DM + nt * 16 * DM + kk * 32, DM, lane);
          acc = wmma_bf16(af[kk], bf, acc);
        }
        int ecol = nt * 16 + (lane & 15);
        float bias = biasS[o * DD + ecol];
#pragma unroll
        for (int j = 0; j < 8; ++j) {
          int row = mi * 16 + j + ((lane >> 4) << 3);
          float v = acc[j] + bias;
          if (o == 0)      qs[row * DD + ecol] = (__bf16)(v * SCALE);
          else if (o == 1) ks[row * DD + ecol] = (__bf16)v;
          else             vT[ecol * MPAD + row] = (__bf16)v;
        }
      }
    }
  }
  __syncthreads();   // also retires xs/wt before Tband/Sband overlay is written

  // ---------------- Phase 2: attention, one 16-row M-tile at a time ----------------
  for (int mi = 0; mi < 13; ++mi) {
    const int row0 = mi * 16;
    v16bf aq = frag_bf(qs + row0 * DD, DD, lane);   // full K=32 q fragment (pre-scaled)

    // T band: (q/sqrt(dm)) @ Eu^T  -> Tband[16][RELP]
    for (int nj = wave; nj < 25; nj += NWAVE) {
      v16bf be = frag_bf(Eus + nj * 16 * DD, DD, lane);
      v8f c = {};
      c = wmma_bf16(aq, be, c);
#pragma unroll
      for (int j = 0; j < 8; ++j) {
        int r = j + ((lane >> 4) << 3);
        Tband[r * RELP + nj * 16 + (lane & 15)] = c[j];
      }
    }
    __syncthreads();

    // qk^T + Srel gather + mask -> Sband
    for (int nj = wave; nj < 13; nj += NWAVE) {
      v16bf bk_ = frag_bf(ks + nj * 16 * DD, DD, lane);
      v8f c = {};
      c = wmma_bf16(aq, bk_, c);
#pragma unroll
      for (int j = 0; j < 8; ++j) {
        int r = j + ((lane >> 4) << 3);
        int l = row0 + r;
        int m = nj * 16 + (lane & 15);
        float s;
        if (l < LL && m < LL)
          s = c[j] + Tband[r * RELP + (199 + m - l)] + mask[l * LL + m] * MASKNEG;
        else
          s = -1e30f;
        Sband[r * MPAD + m] = s;
      }
    }
    for (int idx = tid; idx < 16 * 16; idx += 128) {  // pad key cols 208..223
      int r = idx >> 4;
      Sband[r * MPAD + 208 + (idx & 15)] = -1e30f;
    }
    __syncthreads();

    // row softmax statistics
    if (tid < 16) {
      float mx = -3.4e38f;
      for (int c = 0; c < MPAD; ++c) mx = fmaxf(mx, Sband[tid * MPAD + c]);
      float sum = 0.f;
      for (int c = 0; c < MPAD; ++c) sum += __expf(Sband[tid * MPAD + c] - mx);
      red[tid] = mx;
      red[16 + tid] = 1.0f / sum;
    }
    __syncthreads();

    // normalize, write attn
    for (int idx = tid; idx < 16 * MPAD; idx += 128) {
      int r = idx / MPAD, m = idx % MPAD;
      float p = __expf(Sband[idx] - red[r]) * red[16 + r];
      Sband[idx] = p;
      int l = row0 + r;
      if (l < LL && m < LL) attn[((a * NUM_HEADS + h) * LL + l) * LL + m] = p;
    }
    __syncthreads();

    // out tile = P @ v   (waves 0,1 handle the two 16-wide d tiles)
    if (wave < 2) {
      const int nt = wave;
      v8f acc = {};
#pragma unroll
      for (int kk = 0; kk < 7; ++kk) {
        v16bf pa = frag_f32(Sband + kk * 32, MPAD, lane);
        v16bf bv_ = frag_bf(vT + nt * 16 * MPAD + kk * 32, MPAD, lane);
        acc = wmma_bf16(pa, bv_, acc);
      }
#pragma unroll
      for (int j = 0; j < 8; ++j) {
        int r = j + ((lane >> 4) << 3);
        int l = row0 + r;
        if (l < LL)
          out[(a * LL + l) * DM + h * DD + nt * 16 + (lane & 15)] = acc[j];
      }
    }
    __syncthreads();
  }
}

extern "C" void kernel_launch(void* const* d_in, const int* in_sizes, int n_in,
                              void* d_out, int out_size, void* d_ws, size_t ws_size,
                              hipStream_t stream) {
  const float* inp  = (const float*)d_in[0];
  const float* mask = (const float*)d_in[1];
  const float* Wq   = (const float*)d_in[2];
  const float* bq   = (const float*)d_in[3];
  const float* Wk   = (const float*)d_in[4];
  const float* bk   = (const float*)d_in[5];
  const float* Wv   = (const float*)d_in[6];
  const float* bv   = (const float*)d_in[7];
  const float* E    = (const float*)d_in[8];
  float* out  = (float*)d_out;
  float* attn = out + (size_t)NUM_AREAS * LL * DM;   // outputs concatenated: out, attn

  dim3 grid(NUM_AREAS * NUM_HEADS);
  dim3 block(128);
  compress_attn_kernel<<<grid, block, 0, stream>>>(inp, mask, Wq, bq, Wk, bk, Wv, bv, E,
                                                   out, attn);
}
